// RowSelfAttention_4277787427284
// MI455X (gfx1250) — matrix operand order repaired and checked by reading, not compiled, not run on hardware
//
#include <hip/hip_runtime.h>
#include <math.h>

// ---------------------------------------------------------------------------
// Types for WMMA fragments (gfx1250, wave32)
// ---------------------------------------------------------------------------
typedef __attribute__((ext_vector_type(16))) __bf16        bf16x16;
typedef __attribute__((ext_vector_type(8)))  float         f32x8;
typedef __attribute__((ext_vector_type(4)))  unsigned int  u32x4;
typedef __attribute__((ext_vector_type(2)))  unsigned int  u32x2;
typedef __attribute__((ext_vector_type(4)))  float         f32x4;
typedef __attribute__((ext_vector_type(4)))  int           i32x4;

union FragAB { bf16x16 v; u32x4 q[2]; };

#define LDSW 40   // LDS row stride in bf16 elements (keeps 16B chunk alignment)

#define AS1 __attribute__((address_space(1)))
#define AS3 __attribute__((address_space(3)))

// 16-byte global -> LDS copy. Prefer CDNA5 async DMA (ASYNCcnt-tracked,
// no VGPR staging, overlaps compute); fall back to a sync copy.
// Builtin prototype (from hipcc diagnostic): (v4i AS1*, v4i AS3*, Imm off, Imm cpol)
__device__ __forceinline__ void async_ld16(const void* g, void* l) {
#if __has_builtin(__builtin_amdgcn_global_load_async_to_lds_b128)
  __builtin_amdgcn_global_load_async_to_lds_b128(
      (AS1 i32x4*)g, (AS3 i32x4*)l, 0, 0);
#else
  *(u32x4*)l = *(const u32x4*)g;
#endif
}

__device__ __forceinline__ void wait_async0() {
#if __has_builtin(__builtin_amdgcn_s_wait_asynccnt)
  __builtin_amdgcn_s_wait_asynccnt(0);
#else
  asm volatile("s_wait_asynccnt 0x0" ::: "memory");
#endif
}

// float -> bf16 bits, round-to-nearest-even
__device__ __forceinline__ unsigned short f2bf(float f) {
  unsigned u = __float_as_uint(f);
  unsigned rnd = 0x7fffu + ((u >> 16) & 1u);
  return (unsigned short)((u + rnd) >> 16);
}

// A-fragment (16x32 bf16, M x K): lane half=lane/16, row=lane%16.
// elems 0-7 -> K = 8*half + 0..7 ; elems 8-15 -> K = 16 + 8*half + 0..7
__device__ __forceinline__ void load_frag_a(FragAB& f, const unsigned short* lds,
                                            int row, int half) {
  f.q[0] = *(const u32x4*)&lds[row * LDSW + 8 * half];
  f.q[1] = *(const u32x4*)&lds[row * LDSW + 16 + 8 * half];
}

// B-fragment (32x16 bf16, K x N) from an N-major (N x K) LDS tile:
// lane n = lane%16, elems 0-7 -> K = 8*half + 0..7 ; elems 8-15 -> K = 16 + 8*half + 0..7
__device__ __forceinline__ void load_frag_b(FragAB& f, const unsigned short* lds,
                                            int row, int half) {
  f.q[0] = *(const u32x4*)&lds[row * LDSW + 8 * half];
  f.q[1] = *(const u32x4*)&lds[row * LDSW + 16 + 8 * half];
}

// ---------------------------------------------------------------------------
// fp32 -> bf16 bulk convert (n must be a multiple of 4)
// ---------------------------------------------------------------------------
__global__ __launch_bounds__(256) void cvt_kernel(const float* __restrict__ src,
                                                  unsigned short* __restrict__ dst,
                                                  int n) {
  int i = (blockIdx.x * 256 + threadIdx.x) * 4;
  if (i >= n) return;
  f32x4 f = *(const f32x4*)&src[i];
  u32x2 o;
  o.x = (unsigned)f2bf(f.x) | ((unsigned)f2bf(f.y) << 16);
  o.y = (unsigned)f2bf(f.z) | ((unsigned)f2bf(f.w) << 16);
  *(u32x2*)&dst[i] = o;
}

// ---------------------------------------------------------------------------
// C[M,N] = (A[M,K] @ B[N,K]^T + bias[N]) * scale * (1 - qmask[m])
// block tile 128x128, 8 waves of 32(M)x64(N), K-step 32,
// async global->LDS with double buffering.
// ---------------------------------------------------------------------------
__global__ __launch_bounds__(256) void gemm_nt_kernel(
    const unsigned short* __restrict__ A, const unsigned short* __restrict__ B,
    const float* __restrict__ bias,
    float* __restrict__ outF, unsigned short* __restrict__ outB,
    int N, int K, int lda, int ldb,
    float scale, const unsigned char* __restrict__ qmask) {
  __shared__ unsigned short sA[2][128 * LDSW];
  __shared__ unsigned short sB[2][128 * LDSW];
  const int tid = threadIdx.x;
  const int n0 = blockIdx.x * 128, m0 = blockIdx.y * 128;
  const int w = tid >> 5, lane = tid & 31, half = lane >> 4, lr = lane & 15;
  const int wm = (w & 3) * 32, wn = (w >> 2) * 64;

  // this thread's two 8-bf16 chunks of the 128x32 tile
  const int r0 = tid >> 2, r1 = r0 + 64, col = (tid & 3) << 3;
  const unsigned short* gA0 = A + (size_t)(m0 + r0) * lda + col;
  const unsigned short* gA1 = A + (size_t)(m0 + r1) * lda + col;
  const unsigned short* gB0 = B + (size_t)(n0 + r0) * ldb + col;
  const unsigned short* gB1 = B + (size_t)(n0 + r1) * ldb + col;
  const int ls0 = r0 * LDSW + col, ls1 = r1 * LDSW + col;

  f32x8 acc[2][4];
#pragma unroll
  for (int mt = 0; mt < 2; ++mt)
#pragma unroll
    for (int nt = 0; nt < 4; ++nt) acc[mt][nt] = (f32x8)0.0f;

  const int T = K >> 5;
  // prologue: tile 0 -> buffer 0
  async_ld16(gA0, &sA[0][ls0]);
  async_ld16(gA1, &sA[0][ls1]);
  async_ld16(gB0, &sB[0][ls0]);
  async_ld16(gB1, &sB[0][ls1]);

  for (int t = 0; t < T; ++t) {
    wait_async0();        // tile t resident in LDS
    __syncthreads();      // visible everywhere; tile t-1 reads all complete
    if (t + 1 < T) {      // stream tile t+1 while computing tile t
      const int kc = (t + 1) << 5, nb = (t + 1) & 1;
      async_ld16(gA0 + kc, &sA[nb][ls0]);
      async_ld16(gA1 + kc, &sA[nb][ls1]);
      async_ld16(gB0 + kc, &sB[nb][ls0]);
      async_ld16(gB1 + kc, &sB[nb][ls1]);
    }
    const int b = t & 1;
    FragAB fa[2], fb[4];
#pragma unroll
    for (int mt = 0; mt < 2; ++mt) load_frag_a(fa[mt], sA[b], wm + mt * 16 + lr, half);
#pragma unroll
    for (int nt = 0; nt < 4; ++nt) load_frag_b(fb[nt], sB[b], wn + nt * 16 + lr, half);
#pragma unroll
    for (int mt = 0; mt < 2; ++mt)
#pragma unroll
      for (int nt = 0; nt < 4; ++nt)
        acc[mt][nt] = __builtin_amdgcn_wmma_f32_16x16x32_bf16(
            false, fa[mt].v, false, fb[nt].v, (short)0, acc[mt][nt], false, false);
  }

#pragma unroll
  for (int mt = 0; mt < 2; ++mt)
#pragma unroll
    for (int nt = 0; nt < 4; ++nt) {
      int gn = n0 + wn + nt * 16 + lr;
      float b = bias[gn];
#pragma unroll
      for (int v = 0; v < 8; ++v) {
        int gm = m0 + wm + mt * 16 + 8 * half + v;
        float val = (acc[mt][nt][v] + b) * scale;
        if (qmask) val *= (qmask[gm] ? 0.0f : 1.0f);
        if (outB) outB[(size_t)gm * N + gn] = f2bf(val);
        else      outF[(size_t)gm * N + gn] = val;
      }
    }
}

// ---------------------------------------------------------------------------
// scores[h,i,j] = sum_{r<128,d<64} Q[r,i,h,d] * K[r,j,h,d]
// per-head 512x512 output, reduction dim 8192 = (r,d); async double-buffered
// ---------------------------------------------------------------------------
__global__ __launch_bounds__(256) void scores_kernel(
    const unsigned short* __restrict__ Q, const unsigned short* __restrict__ Kb,
    float* __restrict__ S) {
  __shared__ unsigned short sA[2][128 * LDSW];
  __shared__ unsigned short sB[2][128 * LDSW];
  const int tid = threadIdx.x;
  const int j0 = blockIdx.x * 128, i0 = blockIdx.y * 128, h = blockIdx.z;
  const int w = tid >> 5, lane = tid & 31, half = lane >> 4, lr = lane & 15;
  const int wm = (w & 3) * 32, wn = (w >> 2) * 64;

  const int r0 = tid >> 2, r1 = r0 + 64, col = (tid & 3) << 3;
  // fixed per-thread parts; tile t adds off(t) = (t/2)*512*768 + (t%2)*32
  const unsigned short* gA0 = Q  + (size_t)(i0 + r0) * 768 + (size_t)h * 64 + col;
  const unsigned short* gA1 = Q  + (size_t)(i0 + r1) * 768 + (size_t)h * 64 + col;
  const unsigned short* gB0 = Kb + (size_t)(j0 + r0) * 768 + (size_t)h * 64 + col;
  const unsigned short* gB1 = Kb + (size_t)(j0 + r1) * 768 + (size_t)h * 64 + col;
  const int ls0 = r0 * LDSW + col, ls1 = r1 * LDSW + col;

  f32x8 acc[2][4];
#pragma unroll
  for (int mt = 0; mt < 2; ++mt)
#pragma unroll
    for (int nt = 0; nt < 4; ++nt) acc[mt][nt] = (f32x8)0.0f;

  const int T = 256;   // 128 rows x 2 d-chunks
  async_ld16(gA0, &sA[0][ls0]);
  async_ld16(gA1, &sA[0][ls1]);
  async_ld16(gB0, &sB[0][ls0]);
  async_ld16(gB1, &sB[0][ls1]);

  for (int t = 0; t < T; ++t) {
    wait_async0();
    __syncthreads();
    if (t + 1 < T) {
      const int tn = t + 1, nb = tn & 1;
      const size_t off = (size_t)(tn >> 1) * (512 * 768) + (size_t)(tn & 1) * 32;
      async_ld16(gA0 + off, &sA[nb][ls0]);
      async_ld16(gA1 + off, &sA[nb][ls1]);
      async_ld16(gB0 + off, &sB[nb][ls0]);
      async_ld16(gB1 + off, &sB[nb][ls1]);
    }
    const int b = t & 1;
    FragAB fa[2], fb[4];
#pragma unroll
    for (int mt = 0; mt < 2; ++mt) load_frag_a(fa[mt], sA[b], wm + mt * 16 + lr, half);
#pragma unroll
    for (int nt = 0; nt < 4; ++nt) load_frag_b(fb[nt], sB[b], wn + nt * 16 + lr, half);
#pragma unroll
    for (int mt = 0; mt < 2; ++mt)
#pragma unroll
      for (int nt = 0; nt < 4; ++nt)
        acc[mt][nt] = __builtin_amdgcn_wmma_f32_16x16x32_bf16(
            false, fa[mt].v, false, fb[nt].v, (short)0, acc[mt][nt], false, false);
  }

#pragma unroll
  for (int mt = 0; mt < 2; ++mt)
#pragma unroll
    for (int nt = 0; nt < 4; ++nt) {
      int gj = j0 + wn + nt * 16 + lr;
#pragma unroll
      for (int v = 0; v < 8; ++v) {
        int gi = i0 + wm + mt * 16 + 8 * half + v;
        S[((size_t)h * 512 + gi) * 512 + gj] = acc[mt][nt][v];
      }
    }
}

// ---------------------------------------------------------------------------
// softmax over j (512) with key mask; probs written bf16
// ---------------------------------------------------------------------------
__global__ __launch_bounds__(256) void softmax_kernel(
    const float* __restrict__ S, const unsigned char* __restrict__ mask,
    unsigned short* __restrict__ P) {
  const int row = blockIdx.x;                 // h*512 + i
  const float* s = S + (size_t)row * 512;
  unsigned short* p = P + (size_t)row * 512;
  __shared__ float red[256];
  const int t = threadIdx.x;
  float v0 = mask[t]       ? -10000.0f : s[t];
  float v1 = mask[t + 256] ? -10000.0f : s[t + 256];
  red[t] = fmaxf(v0, v1);
  __syncthreads();
  for (int off = 128; off > 0; off >>= 1) {
    if (t < off) red[t] = fmaxf(red[t], red[t + off]);
    __syncthreads();
  }
  float mx = red[0];
  __syncthreads();
  float e0 = __expf(v0 - mx), e1 = __expf(v1 - mx);
  red[t] = e0 + e1;
  __syncthreads();
  for (int off = 128; off > 0; off >>= 1) {
    if (t < off) red[t] += red[t + off];
    __syncthreads();
  }
  float inv = 1.0f / red[0];
  p[t]       = f2bf(e0 * inv);
  p[t + 256] = f2bf(e1 * inv);
}

// ---------------------------------------------------------------------------
// ctx[r,i,h,d] = sum_j P[h,i,j] * V[r,j,h,d]
// per (r,h): 512(i) x 64(d), K = j = 512. Probs tile async double-buffered;
// V tile software-pipelined (global load issued before the async wait so its
// latency hides), then scattered transposed into LDS after the barrier.
// ---------------------------------------------------------------------------
__global__ __launch_bounds__(256) void ctx_kernel(
    const unsigned short* __restrict__ P, const unsigned short* __restrict__ V,
    unsigned short* __restrict__ CTX) {
  __shared__ unsigned short sA[2][128 * LDSW];
  __shared__ unsigned short sB[2][64 * LDSW];
  const int tid = threadIdx.x;
  const int i0 = blockIdx.x * 128, h = blockIdx.y, r = blockIdx.z;
  const int w = tid >> 5, lane = tid & 31, half = lane >> 4, lr = lane & 15;
  const int wm = (w & 3) * 32, wn = (w >> 2) * 32;

  const unsigned short* Ph = P + (size_t)h * 512 * 512;
  const unsigned short* Vb = V + (size_t)r * 512 * 768 + (size_t)h * 64;

  // probs tile chunks (128 i x 32 j)
  const int r0 = tid >> 2, r1 = r0 + 64, col = (tid & 3) << 3;
  const unsigned short* gA0 = Ph + (size_t)(i0 + r0) * 512 + col;
  const unsigned short* gA1 = Ph + (size_t)(i0 + r1) * 512 + col;
  const int ls0 = r0 * LDSW + col, ls1 = r1 * LDSW + col;
  // V tile: this thread reads 8 contiguous d at (j0+jj, dd), scatters to sB[d][j]
  const int jj = tid >> 3, dd = (tid & 7) << 3;
  const unsigned short* gV = Vb + (size_t)jj * 768 + dd;

  f32x8 acc[2][2];
#pragma unroll
  for (int mt = 0; mt < 2; ++mt)
#pragma unroll
    for (int nt = 0; nt < 2; ++nt) acc[mt][nt] = (f32x8)0.0f;

  const int T = 16;   // 512 / 32
  union { u32x4 q; unsigned short s[8]; } uv;

  // prologue: tile 0
  async_ld16(gA0, &sA[0][ls0]);
  async_ld16(gA1, &sA[0][ls1]);
  uv.q = *(const u32x4*)gV;
#pragma unroll
  for (int x = 0; x < 8; ++x) sB[0][(dd + x) * LDSW + jj] = uv.s[x];

  for (int t = 0; t < T; ++t) {
    union { u32x4 q; unsigned short s[8]; } uvn;
    if (t + 1 < T)   // start V(t+1) global load early; latency hides under waits
      uvn.q = *(const u32x4*)(gV + (size_t)(t + 1) * 32 * 768);
    wait_async0();
    __syncthreads();
    if (t + 1 < T) {
      const int nb = (t + 1) & 1, kc = (t + 1) << 5;
      async_ld16(gA0 + kc, &sA[nb][ls0]);
      async_ld16(gA1 + kc, &sA[nb][ls1]);
#pragma unroll
      for (int x = 0; x < 8; ++x) sB[nb][(dd + x) * LDSW + jj] = uvn.s[x];
    }
    const int b = t & 1;
    FragAB fa[2], fb[2];
#pragma unroll
    for (int mt = 0; mt < 2; ++mt) load_frag_a(fa[mt], sA[b], wm + mt * 16 + lr, half);
#pragma unroll
    for (int nt = 0; nt < 2; ++nt) load_frag_b(fb[nt], sB[b], wn + nt * 16 + lr, half);
#pragma unroll
    for (int mt = 0; mt < 2; ++mt)
#pragma unroll
      for (int nt = 0; nt < 2; ++nt)
        acc[mt][nt] = __builtin_amdgcn_wmma_f32_16x16x32_bf16(
            false, fa[mt].v, false, fb[nt].v, (short)0, acc[mt][nt], false, false);
  }

#pragma unroll
  for (int mt = 0; mt < 2; ++mt)
#pragma unroll
    for (int nt = 0; nt < 2; ++nt) {
      int gd = wn + nt * 16 + lr;
#pragma unroll
      for (int v = 0; v < 8; ++v) {
        int gi = i0 + wm + mt * 16 + 8 * half + v;
        CTX[(size_t)(r * 512 + gi) * 768 + h * 64 + gd] = f2bf(acc[mt][nt][v]);
      }
    }
}

// ---------------------------------------------------------------------------
// Host-side orchestration
// ---------------------------------------------------------------------------
extern "C" void kernel_launch(void* const* d_in, const int* in_sizes, int n_in,
                              void* d_out, int out_size, void* d_ws, size_t ws_size,
                              hipStream_t stream) {
  (void)in_sizes; (void)n_in; (void)out_size; (void)ws_size;
  const float*          X    = (const float*)d_in[0];
  const unsigned char*  mask = (const unsigned char*)d_in[1];   // (1,128,512) bool
  const float* Wq = (const float*)d_in[2]; const float* bq = (const float*)d_in[3];
  const float* Wk = (const float*)d_in[4]; const float* bk = (const float*)d_in[5];
  const float* Wv = (const float*)d_in[6]; const float* bv = (const float*)d_in[7];
  const float* Wo = (const float*)d_in[8]; const float* bo = (const float*)d_in[9];
  float* out = (float*)d_out;

  const size_t TOK = 65536, H = 768;      // 128*512 tokens
  unsigned short* p   = (unsigned short*)d_ws;
  unsigned short* Xb  = p; p += TOK * H;
  unsigned short* Wqb = p; p += H * H;
  unsigned short* Wkb = p; p += H * H;
  unsigned short* Wvb = p; p += H * H;
  unsigned short* Wob = p; p += H * H;
  unsigned short* Qb  = p; p += TOK * H;
  unsigned short* Kb  = p; p += TOK * H;
  unsigned short* Vb  = p; p += TOK * H;
  float*          S   = (float*)p;  p = (unsigned short*)(S + (size_t)12 * 512 * 512);
  unsigned short* Pb  = p; p += (size_t)12 * 512 * 512;
  unsigned short* CTXb = p;

  const float scaleQ = 0.125f / sqrtf(128.0f);   // HEAD_DIM^-0.5 / sqrt(R)

  // fp32 -> bf16 conversions
  cvt_kernel<<<(TOK * H) / 1024, 256, 0, stream>>>(X, Xb, (int)(TOK * H));
  cvt_kernel<<<(H * H) / 1024, 256, 0, stream>>>(Wq, Wqb, (int)(H * H));
  cvt_kernel<<<(H * H) / 1024, 256, 0, stream>>>(Wk, Wkb, (int)(H * H));
  cvt_kernel<<<(H * H) / 1024, 256, 0, stream>>>(Wv, Wvb, (int)(H * H));
  cvt_kernel<<<(H * H) / 1024, 256, 0, stream>>>(Wo, Wob, (int)(H * H));

  // Q/K/V projections (Q fused with scaling + query padding mask)
  dim3 gProj(768 / 128, 65536 / 128);
  gemm_nt_kernel<<<gProj, 256, 0, stream>>>(Xb, Wqb, bq, nullptr, Qb,
                                            768, 768, 768, 768, scaleQ, mask);
  gemm_nt_kernel<<<gProj, 256, 0, stream>>>(Xb, Wkb, bk, nullptr, Kb,
                                            768, 768, 768, 768, 1.0f, nullptr);
  gemm_nt_kernel<<<gProj, 256, 0, stream>>>(Xb, Wvb, bv, nullptr, Vb,
                                            768, 768, 768, 768, 1.0f, nullptr);

  // tied-row scores, softmax, context
  scores_kernel<<<dim3(4, 4, 12), 256, 0, stream>>>(Qb, Kb, S);
  softmax_kernel<<<12 * 512, 256, 0, stream>>>(S, mask, Pb);
  ctx_kernel<<<dim3(4, 12, 128), 256, 0, stream>>>(Pb, Vb, CTXb);

  // output projection (fp32 out)
  gemm_nt_kernel<<<gProj, 256, 0, stream>>>(CTXb, Wob, bo, out, nullptr,
                                            768, 768, 768, 768, 1.0f, nullptr);
}
